// LinRec_Model_bert_34299608826431
// MI455X (gfx1250) — compile-verified, matrix operand order checked by reading, not run on hardware
//
#include <hip/hip_runtime.h>
#include <hip/hip_bf16.h>
#include <stdint.h>

typedef __attribute__((ext_vector_type(16))) _Float16 v16h;
typedef __attribute__((ext_vector_type(8)))  _Float16 v8h;
typedef __attribute__((ext_vector_type(4)))  _Float16 v4h;
typedef __attribute__((ext_vector_type(8)))  float    v8f;
typedef __attribute__((ext_vector_type(4)))  float    v4f;

#define BM 128
#define BN 128
#define BK 32
#define PADK 40   // halves per LDS row: 32 + 8 pad; rows stay 16B aligned

// ---------------------------------------------------------------------------
// Generic f32-in / f32-out GEMM, f16 WMMA compute, f32 accumulate.
//   C[m,n] = act( sum_k A[m,k]*B[k,n] + bias[n] )
//   A addressed as A[(long)m*sAm + (long)k*sAk]; AROW => sAk==1
//   B addressed as B[(long)k*sBk + (long)n*sBn]; BROW => sBn==1
// Block: 256 threads (8 wave32), tile 128x128, K-step 32.
// Waves arranged 4(m) x 2(n); each wave computes 32x64 => 8 WMMA accumulators.
// Grid: blockIdx.x = N tile (dispatch-fastest) so consecutive blocks reuse the
// same A strip in the 192MB L2, and B becomes L2-resident across M strips.
// Layout flags are template params: one staging path per instantiation keeps
// VGPR pressure low. Edge tiles batch loads through a temp array.
// ---------------------------------------------------------------------------
template <bool AROW, bool BROW>
__global__ __launch_bounds__(256) void gemm_f16_wmma(
    const float* __restrict__ A, long sAm, long sAk,
    const float* __restrict__ B, long sBk, long sBn,
    const float* __restrict__ bias,
    float* __restrict__ C, int ldc,
    int M, int N, int K, int doRelu)
{
    __shared__ __align__(16) _Float16 As[BM * PADK];
    __shared__ __align__(16) _Float16 Bs[BN * PADK];

    const int tid  = threadIdx.x;
    const int wave = tid >> 5;
    const int lane = tid & 31;
    const int n0 = blockIdx.x * BN;   // N tiles iterate fastest (L2 A-strip reuse)
    const int m0 = blockIdx.y * BM;

    const int wr = (wave >> 1) * 32;  // wave row offset in tile
    const int wc = (wave & 1) * 64;   // wave col offset in tile

    v8f acc[2][4];
#pragma unroll
    for (int i = 0; i < 2; ++i)
#pragma unroll
        for (int j = 0; j < 4; ++j) acc[i][j] = (v8f){};

    const bool aMfull = (m0 + BM <= M);
    const bool bNfull = (n0 + BN <= N);

    for (int k0 = 0; k0 < K; k0 += BK) {
        const bool kFull = (k0 + BK <= K);
        __syncthreads();

        // ================= stage A tile (128m x 32k) =================
        if (aMfull && kFull) {
            if (AROW) {
#pragma unroll
                for (int i = 0; i < 4; ++i) {
                    int vid = tid + i * 256;
                    int row = vid >> 3, c4 = (vid & 7) * 4;
                    v4f f = *(const v4f*)(A + (long)(m0 + row) * sAm + (k0 + c4));
                    v4h h = { (_Float16)f[0], (_Float16)f[1],
                              (_Float16)f[2], (_Float16)f[3] };
                    *(v4h*)(&As[row * PADK + c4]) = h;
                }
            } else { // col-major: contiguous along m
#pragma unroll
                for (int i = 0; i < 4; ++i) {
                    int vid = tid + i * 256;
                    int kcol = vid >> 5, m4 = (vid & 31) * 4;
                    v4f f = *(const v4f*)(A + (long)(k0 + kcol) * sAk + (m0 + m4));
#pragma unroll
                    for (int j = 0; j < 4; ++j)
                        As[(m4 + j) * PADK + kcol] = (_Float16)f[j];
                }
            }
        } else {
            // edge: batch clamped loads into temps (mask applied by multiply),
            // then convert + store. Keeps loads un-predicated and in flight.
            float tmp[16];
#pragma unroll
            for (int i = 0; i < 16; ++i) {
                int e = tid + i * 256;
                int m, kk;
                if (AROW) { kk = e & 31;  m  = e >> 5; }
                else      { m  = e & 127; kk = e >> 7; }
                int gm = m0 + m, gk = k0 + kk;
                int cm = gm < M ? gm : (M - 1);
                int ck = gk < K ? gk : (K - 1);
                float msk = (gm < M && gk < K) ? 1.0f : 0.0f;
                tmp[i] = A[(long)cm * sAm + (long)ck * sAk] * msk;
            }
#pragma unroll
            for (int i = 0; i < 16; ++i) {
                int e = tid + i * 256;
                int m, kk;
                if (AROW) { kk = e & 31;  m  = e >> 5; }
                else      { m  = e & 127; kk = e >> 7; }
                As[m * PADK + kk] = (_Float16)tmp[i];
            }
        }

        // ============ stage B tile (32k x 128n), stored Bs[n][k] ============
        if (bNfull && kFull) {
            if (BROW) {
#pragma unroll
                for (int i = 0; i < 4; ++i) {
                    int vid = tid + i * 256;
                    int krow = vid >> 5, n4 = (vid & 31) * 4;
                    v4f f = *(const v4f*)(B + (long)(k0 + krow) * sBk + (n0 + n4));
#pragma unroll
                    for (int j = 0; j < 4; ++j)
                        Bs[(n4 + j) * PADK + krow] = (_Float16)f[j];
                }
            } else { // col-major: contiguous along k
#pragma unroll
                for (int i = 0; i < 4; ++i) {
                    int vid = tid + i * 256;
                    int nrow = vid >> 3, k4 = (vid & 7) * 4;
                    v4f f = *(const v4f*)(B + (long)(n0 + nrow) * sBn + (k0 + k4));
                    v4h h = { (_Float16)f[0], (_Float16)f[1],
                              (_Float16)f[2], (_Float16)f[3] };
                    *(v4h*)(&Bs[nrow * PADK + k4]) = h;
                }
            }
        } else {
            float tmp[16];
#pragma unroll
            for (int i = 0; i < 16; ++i) {
                int e = tid + i * 256;
                int n, kk;
                if (BROW) { n  = e & 127; kk = e >> 7; }
                else      { kk = e & 31;  n  = e >> 5; }
                int gn = n0 + n, gk = k0 + kk;
                int cn = gn < N ? gn : (N - 1);
                int ck = gk < K ? gk : (K - 1);
                float msk = (gn < N && gk < K) ? 1.0f : 0.0f;
                tmp[i] = B[(long)ck * sBk + (long)cn * sBn] * msk;
            }
#pragma unroll
            for (int i = 0; i < 16; ++i) {
                int e = tid + i * 256;
                int n, kk;
                if (BROW) { n  = e & 127; kk = e >> 7; }
                else      { kk = e & 31;  n  = e >> 5; }
                Bs[n * PADK + kk] = (_Float16)tmp[i];
            }
        }
        __syncthreads();

        // ================= fragments + 8 WMMAs =================
        // CDNA5 16-bit 16x32 layout: lanes 0-15 row r, K koff..+7 / koff+16..+23
        const int r    = lane & 15;
        const int koff = (lane >> 4) << 3;

        v16h afr[2], bfr[4];
#pragma unroll
        for (int t = 0; t < 2; ++t) {
            const _Float16* ap = &As[(wr + t * 16 + r) * PADK + koff];
            v8h lo = *(const v8h*)(ap), hi = *(const v8h*)(ap + 16);
#pragma unroll
            for (int j = 0; j < 8; ++j) { afr[t][j] = lo[j]; afr[t][j + 8] = hi[j]; }
        }
#pragma unroll
        for (int t = 0; t < 4; ++t) {
            const _Float16* bp = &Bs[(wc + t * 16 + r) * PADK + koff];
            v8h lo = *(const v8h*)(bp), hi = *(const v8h*)(bp + 16);
#pragma unroll
            for (int j = 0; j < 8; ++j) { bfr[t][j] = lo[j]; bfr[t][j + 8] = hi[j]; }
        }

#pragma unroll
        for (int t = 0; t < 2; ++t)
#pragma unroll
            for (int u = 0; u < 4; ++u)
                acc[t][u] = __builtin_amdgcn_wmma_f32_16x16x32_f16(
                    false, afr[t], false, bfr[u], (short)0, acc[t][u], false, false);
    }

    // ===== epilogue: C/D layout => N = lane&15, rows e + (lane>=16 ? 8:0) =====
    const int rl  = lane & 15;
    const int rhi = (lane >> 4) << 3;

#pragma unroll
    for (int u = 0; u < 4; ++u) {
        const int cn = n0 + wc + u * 16 + rl;
        if (cn >= N) continue;
        const float bia = bias ? bias[cn] : 0.0f;
#pragma unroll
        for (int t = 0; t < 2; ++t) {
            const int cmBase = m0 + wr + t * 16 + rhi;
#pragma unroll
            for (int ee = 0; ee < 8; ++ee) {
                int cm = cmBase + ee;
                if (cm < M) {
                    float v = acc[t][u][ee] + bia;
                    if (doRelu) v = fmaxf(v, 0.0f);
                    C[(long)cm * ldc + cn] = v;
                }
            }
        }
    }
}

// ---------------------------------------------------------------------------
// Gating: one wave32 per row.  gates[row] = softmax(seq_emb[row] @ gW + gb)
// ---------------------------------------------------------------------------
__global__ __launch_bounds__(256) void gate_kernel(
    const float* __restrict__ seq, const float* __restrict__ gW,
    const float* __restrict__ gb, float* __restrict__ gates, int rows)
{
    const int row  = blockIdx.x * 8 + (threadIdx.x >> 5);
    const int lane = threadIdx.x & 31;
    if (row >= rows) return;
    float a0 = 0.f, a1 = 0.f, a2 = 0.f;
    const float* x = seq + (long)row * 256;
    for (int d = lane; d < 256; d += 32) {
        float s = x[d];
        a0 += s * gW[d * 3 + 0];
        a1 += s * gW[d * 3 + 1];
        a2 += s * gW[d * 3 + 2];
    }
#pragma unroll
    for (int off = 16; off > 0; off >>= 1) {
        a0 += __shfl_xor(a0, off, 32);
        a1 += __shfl_xor(a1, off, 32);
        a2 += __shfl_xor(a2, off, 32);
    }
    if (lane == 0) {
        a0 += gb[0]; a1 += gb[1]; a2 += gb[2];
        float m = fmaxf(a0, fmaxf(a1, a2));
        float e0 = __expf(a0 - m), e1 = __expf(a1 - m), e2 = __expf(a2 - m);
        float inv = 1.0f / (e0 + e1 + e2);
        gates[row * 3 + 0] = e0 * inv;
        gates[row * 3 + 1] = e1 * inv;
        gates[row * 3 + 2] = e2 * inv;
    }
}

// ---------------------------------------------------------------------------
// Row softmax with pre-scale: x = softmax(scale * x), one block per row.
// ---------------------------------------------------------------------------
__global__ __launch_bounds__(256) void softmax_rows(
    float* __restrict__ X, int S, float scale)
{
    __shared__ float red[256];
    float* x = X + (long)blockIdx.x * S;
    const int tid = threadIdx.x;

    float m = -3.4e38f;
    for (int i = tid; i < S; i += 256) m = fmaxf(m, x[i]);
    red[tid] = m; __syncthreads();
    for (int s = 128; s > 0; s >>= 1) {
        if (tid < s) red[tid] = fmaxf(red[tid], red[tid + s]);
        __syncthreads();
    }
    m = red[0] * scale;
    __syncthreads();

    float sum = 0.f;
    for (int i = tid; i < S; i += 256) {
        float e = __expf(x[i] * scale - m);
        x[i] = e; sum += e;
    }
    red[tid] = sum; __syncthreads();
    for (int s = 128; s > 0; s >>= 1) {
        if (tid < s) red[tid] += red[tid + s];
        __syncthreads();
    }
    const float inv = 1.0f / red[0];
    for (int i = tid; i < S; i += 256) x[i] *= inv;
}

// ---------------------------------------------------------------------------
// out[row,0:64] (+)= gates[row,e] * eo[row,0:64]
// ---------------------------------------------------------------------------
__global__ __launch_bounds__(256) void combine_kernel(
    const float* __restrict__ eo, const float* __restrict__ gates,
    int e, float* __restrict__ out, int rows)
{
    int idx = blockIdx.x * 256 + threadIdx.x;
    if (idx >= rows * 64) return;
    int row = idx >> 6;
    float v = gates[row * 3 + e] * eo[idx];
    if (e == 0) out[idx] = v;
    else        out[idx] += v;
}

// ---------------------------------------------------------------------------
extern "C" void kernel_launch(void* const* d_in, const int* in_sizes, int n_in,
                              void* d_out, int out_size, void* d_ws, size_t ws_size,
                              hipStream_t stream) {
    (void)in_sizes; (void)n_in; (void)out_size; (void)ws_size;

    // ---- input unpacking (setup_inputs dict order, lists flattened) ----
    const float* seq_emb = (const float*)d_in[0];
    // d_in[1] = seq (int) — unused by the reference math
    const float* word_embs[3] = { (const float*)d_in[2],
                                  (const float*)d_in[3],
                                  (const float*)d_in[4] };
    const float* gate_W = (const float*)d_in[5];
    const float* gate_b = (const float*)d_in[6];
    // mlp_params: expert e layer l  -> W = d_in[7 + e*12 + 2l], b = +1
    // expert_params: expert e       -> base 43 + e*8 : Wq,bq,Wk,bk,Wv,bv,Wo,bo

    const int DLLM[3] = { 4096, 768, 768 };
    const int VOC[3]  = { 32000, 30522, 32100 };
    const int MLPN[6] = { 1024, 1024, 512, 256, 128, 1000 };
    const int ROWS = 32 * 128;                // B*L = 4096
    const float SCALE = 0.17677669529663687f; // 1/sqrt(32)

    // ---- workspace carve-out ----
    uintptr_t p = (uintptr_t)d_ws;
    auto alloc = [&](size_t bytes) -> float* {
        float* r = (float*)p;
        p += (bytes + 255) & ~(size_t)255;
        return r;
    };
    float* gates  = alloc((size_t)ROWS * 3    * 4);
    float* actA   = alloc((size_t)4096 * 1024 * 4);
    float* actB   = alloc((size_t)4096 * 1024 * 4);
    float* qbuf   = alloc((size_t)ROWS * 256  * 4);
    float* kbuf   = alloc((size_t)1000 * 256  * 4);
    float* vbuf   = alloc((size_t)1000 * 256  * 4);
    float* scores = alloc((size_t)ROWS * 1000 * 4);
    float* attn   = alloc((size_t)ROWS * 256  * 4);
    float* eo     = alloc((size_t)ROWS * 64   * 4);
    float* ping[2] = { actA, actB };

    auto gemm = [&](const float* A, long sAm, long sAk,
                    const float* B, long sBk, long sBn,
                    const float* bias, float* C, int ldc,
                    int M, int N, int K, int relu) {
        dim3 grid((N + BN - 1) / BN, (M + BM - 1) / BM, 1); // x = N tiles
        const bool aRow = (sAk == 1), bRow = (sBn == 1);
        if (aRow && bRow)
            gemm_f16_wmma<true, true><<<grid, 256, 0, stream>>>(
                A, sAm, sAk, B, sBk, sBn, bias, C, ldc, M, N, K, relu);
        else if (aRow && !bRow)
            gemm_f16_wmma<true, false><<<grid, 256, 0, stream>>>(
                A, sAm, sAk, B, sBk, sBn, bias, C, ldc, M, N, K, relu);
        else if (!aRow && bRow)
            gemm_f16_wmma<false, true><<<grid, 256, 0, stream>>>(
                A, sAm, sAk, B, sBk, sBn, bias, C, ldc, M, N, K, relu);
        else
            gemm_f16_wmma<false, false><<<grid, 256, 0, stream>>>(
                A, sAm, sAk, B, sBk, sBn, bias, C, ldc, M, N, K, relu);
    };

    // ---- gating ----
    gate_kernel<<<(ROWS + 7) / 8, 256, 0, stream>>>(seq_emb, gate_W, gate_b,
                                                    gates, ROWS);

    for (int e = 0; e < 3; ++e) {
        const int D = DLLM[e];

        // ---- MLP chain over word_embs[e].T : (D x V) -> ... -> (D x 1000) ----
        const float* Aptr = word_embs[e];
        long sAm = 1, sAk = D;            // transposed view of (V x D) table
        int  Kdim = VOC[e];
        for (int l = 0; l < 6; ++l) {
            const float* W = (const float*)d_in[7 + e * 12 + 2 * l];
            const float* b = (const float*)d_in[7 + e * 12 + 2 * l + 1];
            int Ndim = MLPN[l];
            gemm(Aptr, sAm, sAk, W, (long)Ndim, 1, b,
                 ping[l & 1], Ndim, D, Ndim, Kdim, /*relu=*/1);
            Aptr = ping[l & 1]; sAm = Ndim; sAk = 1; Kdim = Ndim;
        }
        const float* srcAct = actB;       // (D x 1000) row-major; src = its T

        const float* Wq = (const float*)d_in[43 + e * 8 + 0];
        const float* bq = (const float*)d_in[43 + e * 8 + 1];
        const float* Wk = (const float*)d_in[43 + e * 8 + 2];
        const float* bk = (const float*)d_in[43 + e * 8 + 3];
        const float* Wv = (const float*)d_in[43 + e * 8 + 4];
        const float* bv = (const float*)d_in[43 + e * 8 + 5];
        const float* Wo = (const float*)d_in[43 + e * 8 + 6];
        const float* bo = (const float*)d_in[43 + e * 8 + 7];

        // q = seq_emb @ Wq + bq            : (4096 x 256)
        gemm(seq_emb, 256, 1, Wq, 256, 1, bq, qbuf, 256, ROWS, 256, 256, 0);
        // k = src @ Wk + bk  (src = actB.T): (1000 x 256)
        gemm(srcAct, 1, 1000, Wk, 256, 1, bk, kbuf, 256, 1000, 256, D, 0);
        // v = src @ Wv + bv                : (1000 x 256)
        gemm(srcAct, 1, 1000, Wv, 256, 1, bv, vbuf, 256, 1000, 256, D, 0);

        // per-head attention: scores -> softmax -> A@V
        for (int h = 0; h < 8; ++h) {
            // scores[bl,s] = q[bl,h,:] . k[s,h,:]   (M=4096,N=1000,K=32)
            gemm(qbuf + h * 32, 256, 1,
                 kbuf + h * 32, 1, 256,   // B[k][n] = kbuf[n*256 + h*32 + k]
                 nullptr, scores, 1000, ROWS, 1000, 32, 0);
            softmax_rows<<<ROWS, 256, 0, stream>>>(scores, 1000, SCALE);
            // attn[bl,h,:] = scores @ v[:,h,:]      (M=4096,N=32,K=1000)
            gemm(scores, 1000, 1,
                 vbuf + h * 32, 256, 1,
                 nullptr, attn + h * 32, 256, ROWS, 32, 1000, 0);
        }

        // expert output: attn @ Wo + bo : (4096 x 64)
        gemm(attn, 256, 1, Wo, 64, 1, bo, eo, 64, ROWS, 64, 256, 0);

        // gated accumulate into d_out
        combine_kernel<<<(ROWS * 64 + 255) / 256, 256, 0, stream>>>(
            eo, gates, e, (float*)d_out, ROWS);
    }
}
